// AttentionCore_55405078118644
// MI455X (gfx1250) — compile-verified, matrix operand order checked by reading, not run on hardware
//
#include <hip/hip_runtime.h>
#include <math.h>

// ---------------------------------------------------------------------------
// AttentionCore for MI455X (gfx1250, wave32, WMMA, async-to-LDS + TDM).
// Pipeline: f32->bf16 converts (weights pre-transposed) -> WMMA QKV GEMM ->
//           RoPE+split -> sliding-window GQA attention (WMMA QK^T and PV) ->
//           WMMA out GEMM. Matrix math on v_wmma_f32_16x16x32_bf16.
// GEMM tiles staged with GLOBAL_LOAD_ASYNC_TO_LDS_B128 (ASYNCcnt, double
// buffered); attention K window staged with TENSOR_LOAD_TO_LDS (TDM,
// TENSORcnt), using TDM OOB zero-fill for the sequence tail.
// ---------------------------------------------------------------------------

#define S_LEN 2048
#define HID   2048
#define NH    32
#define KVH   8
#define HD    64
#define WIN   128
#define QM    4
#define SM_SCALE 0.125f   // 1/sqrt(64)

typedef __attribute__((ext_vector_type(16))) __bf16 v16bf;
typedef __attribute__((ext_vector_type(8)))  float  v8f;
typedef __attribute__((ext_vector_type(4)))  int    v4i_;
typedef __attribute__((ext_vector_type(8)))  int    v8i_;

union Frag16 {
    v16bf v;
    uint4 q[2];
};

static __device__ __forceinline__ unsigned short f2bf(float f) {
    unsigned int u = __float_as_uint(f);
    u = u + 0x7FFFu + ((u >> 16) & 1u);   // round-to-nearest-even
    return (unsigned short)(u >> 16);
}

// Async copy of 16B global -> LDS (per lane). dsaddr = low 32 bits of the
// generic shared pointer (ISA 10.2: addr[31:0] of a shared flat address is
// the LDS offset). Tracked by ASYNCcnt.
static __device__ __forceinline__ void async_cp16(const void* gptr, void* lptr) {
    unsigned long long ga = (unsigned long long)(size_t)gptr;
    unsigned           la = (unsigned)(size_t)lptr;
    asm volatile("global_load_async_to_lds_b128 %0, %1, off"
                 :: "v"(la), "v"(ga) : "memory");
}
static __device__ __forceinline__ void async_wait0() {
    asm volatile("s_wait_asynccnt 0x0" ::: "memory");
}

// Tensor Data Mover: 2D tile load Global -> LDS (ISA 08_async_tensor 7/8).
// D# group0: count=1 | lds_addr | global_addr[56:0] | type=2.
// D# group1: wg_mask=0, data_size=1 (2B), tensor_dim0/1, tile_dim0/1,
//            tensor_dim0_stride. Rows packed contiguously in LDS.
// Reads past tensor_dim return zero (used for the sequence tail).
static __device__ __forceinline__ void tdm_load_2d_bf16(
    unsigned lds_addr, const void* gptr,
    unsigned tile_d0, unsigned tile_d1,        // tile: elems per row, rows
    unsigned tensor_d0, unsigned tensor_d1,    // OOB extents (elems, rows)
    unsigned d0_stride)                        // row stride in elements
{
    unsigned long long ga = (unsigned long long)(size_t)gptr;
    v4i_ g0;
    g0[0] = 1;                                             // count=1, user D#
    g0[1] = (int)lds_addr;                                 // lds byte address
    g0[2] = (int)(unsigned)ga;                             // global[31:0]
    g0[3] = (int)((unsigned)((ga >> 32) & 0x1FFFFFFull) | (2u << 30)); // [56:32]|type=2
    v8i_ g1;
    g1[0] = 0x00010000;                                    // mask=0, data_size=1 (2B)
    g1[1] = (int)((tensor_d0 & 0xFFFFu) << 16);            // tensor_dim0[15:0]
    g1[2] = (int)(((tensor_d0 >> 16) & 0xFFFFu) | ((tensor_d1 & 0xFFFFu) << 16));
    g1[3] = (int)(((tensor_d1 >> 16) & 0xFFFFu) | ((tile_d0 & 0xFFFFu) << 16));
    g1[4] = (int)(tile_d1 & 0xFFFFu);                      // tile_dim1 (dim2=0)
    g1[5] = (int)d0_stride;                                // dim0_stride[31:0]
    g1[6] = 0;
    g1[7] = 0;
    asm volatile("tensor_load_to_lds %0, %1" :: "s"(g0), "s"(g1) : "memory");
}

// ---------------------------------------------------------------------------
// fp32 -> bf16 bulk conversion (row-major preserved)
// ---------------------------------------------------------------------------
__global__ __launch_bounds__(256) void cvt_f32_bf16_kernel(
    const float* __restrict__ in, unsigned short* __restrict__ out, int n)
{
    int i = blockIdx.x * 256 + threadIdx.x;
    if (i < n) out[i] = f2bf(in[i]);
}

// ---------------------------------------------------------------------------
// fp32 [K][N] -> bf16 [N][K] transposed conversion (32x32 LDS tiles).
// ---------------------------------------------------------------------------
__global__ __launch_bounds__(256) void cvt_transpose_bf16_kernel(
    const float* __restrict__ in, unsigned short* __restrict__ out, int K, int N)
{
    __shared__ float tile[32][33];
    const int n0 = blockIdx.x * 32;
    const int k0 = blockIdx.y * 32;
    const int tx = threadIdx.x & 31;
    const int ty = threadIdx.x >> 5;          // 8 row-groups
    #pragma unroll
    for (int r = ty; r < 32; r += 8)
        tile[r][tx] = in[(size_t)(k0 + r) * N + n0 + tx];
    __syncthreads();
    #pragma unroll
    for (int r = ty; r < 32; r += 8)
        out[(size_t)(n0 + r) * K + k0 + tx] = f2bf(tile[tx][r]);
}

// ---------------------------------------------------------------------------
// bf16 GEMM: C[M,N] = A[M,K] * Bt[N,K]^T + bias[N]  (fp32 accumulate/output)
// Block tile 128x128, K-chunk 32, 8 waves; wave w owns rows [16w,16w+16) and
// all 128 cols (8 WMMA n-tiles). Tiles staged async into a double buffer:
//   wait(cur) -> barrier -> issue async(next) -> WMMAs(cur).
// Fragment layout (16-bit A-matrix 16x32): half h=lane/16, row/col=lane%16,
// K elems {8h..8h+7} U {16+8h..23+8h} -> 2x ds_load_b128.
// ---------------------------------------------------------------------------
#define GBM 128
#define GBN 128
#define GBK 32
#define GLD 40   // padded K stride (elements) for LDS tiles

__global__ __launch_bounds__(256) void gemm_bf16_kernel(
    const unsigned short* __restrict__ A,    // [M][K] bf16
    const unsigned short* __restrict__ Bt,   // [N][K] bf16 (pre-transposed)
    const float* __restrict__ bias,          // [N]
    float* __restrict__ C,                   // [M][N]
    int M, int N, int K)
{
    __shared__ unsigned short As[2][GBM * GLD];
    __shared__ unsigned short Bs[2][GBN * GLD];

    const int tid  = threadIdx.x;
    const int lane = tid & 31;
    const int wave = tid >> 5;
    const int m0 = blockIdx.y * GBM;
    const int n0 = blockIdx.x * GBN;
    const int half = lane >> 4;
    const int lr   = lane & 15;

    // 512 16B slots per tile, 256 threads -> exactly 2 unconditional copies
    // per thread per tile (no lane-varying bounds -> no exec juggling).
    const int r0 = tid >> 2,           s0 = tid & 3;
    const int r1 = (tid + 256) >> 2,   s1 = (tid + 256) & 3;

    auto stage = [&](int kb, int buf) {
        const unsigned short* Ab = &A[(size_t)m0 * K + kb * GBK];
        async_cp16(&Ab[(size_t)r0 * K + s0 * 8], &As[buf][r0 * GLD + s0 * 8]);
        async_cp16(&Ab[(size_t)r1 * K + s1 * 8], &As[buf][r1 * GLD + s1 * 8]);
        const unsigned short* Bb = &Bt[(size_t)n0 * K + kb * GBK];
        async_cp16(&Bb[(size_t)r0 * K + s0 * 8], &Bs[buf][r0 * GLD + s0 * 8]);
        async_cp16(&Bb[(size_t)r1 * K + s1 * 8], &Bs[buf][r1 * GLD + s1 * 8]);
    };

    v8f acc[8];
    #pragma unroll
    for (int t = 0; t < 8; ++t) acc[t] = (v8f){0.f,0.f,0.f,0.f,0.f,0.f,0.f,0.f};

    const int nK = K / GBK;
    stage(0, 0);
    for (int kb = 0; kb < nK; ++kb) {
        const int buf = kb & 1;
        async_wait0();        // own stage(kb) copies landed
        __syncthreads();      // everyone's stage(kb) landed; prior reads done
        if (kb + 1 < nK) stage(kb + 1, buf ^ 1);   // overlaps WMMAs below

        Frag16 a;
        const int arow = wave * 16 + lr;
        a.q[0] = *(const uint4*)&As[buf][arow * GLD + 8 * half];
        a.q[1] = *(const uint4*)&As[buf][arow * GLD + 16 + 8 * half];
        #pragma unroll
        for (int t = 0; t < 8; ++t) {
            Frag16 b;
            const int bcol = t * 16 + lr;
            b.q[0] = *(const uint4*)&Bs[buf][bcol * GLD + 8 * half];
            b.q[1] = *(const uint4*)&Bs[buf][bcol * GLD + 16 + 8 * half];
            acc[t] = __builtin_amdgcn_wmma_f32_16x16x32_bf16(
                false, a.v, false, b.v, (short)0, acc[t], false, false);
        }
    }

    // epilogue: C layout -> lane<16: M=r, lane>=16: M=8+r; N = lane%16
    #pragma unroll
    for (int t = 0; t < 8; ++t) {
        #pragma unroll
        for (int r = 0; r < 8; ++r) {
            int m = m0 + wave * 16 + r + 8 * half;
            int n = n0 + t * 16 + lr;
            C[(size_t)m * N + n] = acc[t][r] + bias[n];
        }
    }
}

// ---------------------------------------------------------------------------
// RoPE + head split (bias already applied by the QKV GEMM epilogue).
// qkv fp32 [S][3072] -> q bf16 [S][32][64] (scaled by SM_SCALE),
// k bf16 [S][8][64], v bf16 [S][8][64]. Rotate-half RoPE.
// ---------------------------------------------------------------------------
__global__ __launch_bounds__(256) void rope_split_kernel(
    const float* __restrict__ qkv,
    const float* __restrict__ cosT, const float* __restrict__ sinT,
    unsigned short* __restrict__ qo, unsigned short* __restrict__ ko,
    unsigned short* __restrict__ vo)
{
    const int s = blockIdx.x;
    for (int idx = threadIdx.x; idx < 48 * 32; idx += 256) {
        const int slot = idx >> 5;   // 0..31 q heads, 32..39 k heads, 40..47 v
        const int d    = idx & 31;
        const int c1 = slot * 64 + d, c2 = c1 + 32;
        float v1 = qkv[(size_t)s * 3072 + c1];
        float v2 = qkv[(size_t)s * 3072 + c2];
        float o1 = v1, o2 = v2;
        if (slot < 40) {
            float c  = cosT[s * 32 + d];
            float sn = sinT[s * 32 + d];
            o1 = v1 * c - v2 * sn;
            o2 = v2 * c + v1 * sn;
        }
        if (slot < 32) {
            o1 *= SM_SCALE; o2 *= SM_SCALE;
            size_t base = ((size_t)s * NH + slot) * HD + d;
            qo[base] = f2bf(o1); qo[base + 32] = f2bf(o2);
        } else if (slot < 40) {
            size_t base = ((size_t)s * KVH + (slot - 32)) * HD + d;
            ko[base] = f2bf(o1); ko[base + 32] = f2bf(o2);
        } else {
            size_t base = ((size_t)s * KVH + (slot - 40)) * HD + d;
            vo[base] = f2bf(o1); vo[base + 32] = f2bf(o2);
        }
    }
}

// ---------------------------------------------------------------------------
// Sliding-window GQA attention with sinks.
// grid = (S/16, KVH); block = 128 (4 waves). Wave w handles q-head kvh*4+w
// for 16 queries. K window staged by the Tensor Data Mover (wave 0 issues;
// TDM ignores EXEC; TENSORcnt waited before the block barrier). Sequence
// tail rows zero-fill via TDM OOB; leading (negative-key) rows are skipped
// and stay fully masked (mask *overwrites* with -inf, so stale LDS there is
// harmless). V is scatter-transposed to [hd][key]. QK^T: 9 key tiles x 2
// WMMAs; softmax row reductions via width-16 shuffles; P round-trips through
// LDS into A-fragment layout for the PV WMMAs (5 k-chunks x 4 hd tiles).
// ---------------------------------------------------------------------------
#define AKEYS 160          // staged keys (window 144 padded to 5x32 chunks)
#define VLD   168          // padded key stride for Vt / Ps

__global__ __launch_bounds__(128) void attn_kernel(
    const unsigned short* __restrict__ Q,    // [S][NH][HD] bf16, pre-scaled
    const unsigned short* __restrict__ Kb,   // [S][KVH][HD] bf16
    const unsigned short* __restrict__ Vb,   // [S][KVH][HD] bf16
    const float* __restrict__ sinks,         // [NH]
    unsigned short* __restrict__ O)          // [S][NH*HD] bf16
{
    __shared__ unsigned short Ks[AKEYS * HD];     // K natural [key][hd]
    __shared__ unsigned short Vt[HD * VLD];       // V transposed [hd][key]
    __shared__ unsigned short Ps[4][16 * VLD];    // per-wave probabilities

    const int tid  = threadIdx.x;
    const int lane = tid & 31;
    const int wave = tid >> 5;
    const int qb   = blockIdx.x;
    const int kvh  = blockIdx.y;
    const int h    = kvh * QM + wave;
    const int kbase = qb * 16 - WIN;
    const int half = lane >> 4;
    const int lr   = lane & 15;

    // ---- stage K via TDM (one wave), V via scatter transpose (all) ----
    const int start = kbase < 0 ? 0 : kbase;   // first real key
    const int skip  = start - kbase;           // masked leading rows
    if (wave == 0) {
        tdm_load_2d_bf16((unsigned)(size_t)&Ks[skip * HD],
                         &Kb[((size_t)start * KVH + kvh) * HD],
                         /*tile d0,d1*/  HD, (unsigned)(AKEYS - skip),
                         /*tensor d0,d1*/HD, (unsigned)(S_LEN - start),
                         /*row stride*/  KVH * HD);
        __builtin_amdgcn_s_wait_tensorcnt(0);
    }
    for (int e = tid; e < AKEYS * (HD / 8); e += 128) {
        int ki = e >> 3, seg = e & 7;
        int key = kbase + ki;
        key = key < 0 ? 0 : (key > S_LEN - 1 ? S_LEN - 1 : key);
        size_t g = ((size_t)key * KVH + kvh) * HD + seg * 8;
        uint4 dv = *(const uint4*)&Vb[g];
        const unsigned short* d16 = (const unsigned short*)&dv;
        #pragma unroll
        for (int j = 0; j < 8; ++j) Vt[(seg * 8 + j) * VLD + ki] = d16[j];
    }
    __syncthreads();

    // ---- Q fragments (A-layout), direct from global ----
    Frag16 aq[2];
    const int qrow = qb * 16 + lr;
    #pragma unroll
    for (int c = 0; c < 2; ++c) {
        size_t g = ((size_t)qrow * NH + h) * HD + c * 32;
        aq[c].q[0] = *(const uint4*)&Q[g + 8 * half];
        aq[c].q[1] = *(const uint4*)&Q[g + 16 + 8 * half];
    }

    // ---- logits: sc[kt] = Q(16x64) * K^T(64x16) for 9 key tiles ----
    v8f sc[9];
    #pragma unroll
    for (int kt = 0; kt < 9; ++kt) {
        v8f c = (v8f){0.f,0.f,0.f,0.f,0.f,0.f,0.f,0.f};
        #pragma unroll
        for (int cc = 0; cc < 2; ++cc) {
            Frag16 b;   // B elem (k=hd, n=key): fixed key per lane, hd contiguous
            const int kl = kt * 16 + lr;
            b.q[0] = *(const uint4*)&Ks[kl * HD + cc * 32 + 8 * half];
            b.q[1] = *(const uint4*)&Ks[kl * HD + cc * 32 + 16 + 8 * half];
            c = __builtin_amdgcn_wmma_f32_16x16x32_bf16(
                false, aq[cc].v, false, b.v, (short)0, c, false, false);
        }
        sc[kt] = c;
    }

    // ---- mask + sink-aware softmax ----
    const float sinkv = sinks[h];
    const float NEG = -__builtin_inff();
    float mx[8], sm[8];
    #pragma unroll
    for (int r = 0; r < 8; ++r) {
        const int m  = r + 8 * half;
        const int qi = qb * 16 + m;
        mx[r] = sinkv;                       // sink participates in the max
        #pragma unroll
        for (int kt = 0; kt < 9; ++kt) {
            const int key = kbase + kt * 16 + lr;
            const bool valid = (key >= 0) && (key <= qi) && (key >= qi - WIN);
            float v = valid ? sc[kt][r] : NEG;
            sc[kt][r] = v;
            mx[r] = fmaxf(mx[r], v);
        }
        #pragma unroll
        for (int st = 1; st < 16; st <<= 1)
            mx[r] = fmaxf(mx[r], __shfl_xor(mx[r], st, 16));
    }
    #pragma unroll
    for (int r = 0; r < 8; ++r) {
        sm[r] = 0.f;
        #pragma unroll
        for (int kt = 0; kt < 9; ++kt) {
            float p = __expf(sc[kt][r] - mx[r]);
            sc[kt][r] = p;
            sm[r] += p;
        }
        #pragma unroll
        for (int st = 1; st < 16; st <<= 1)
            sm[r] += __shfl_xor(sm[r], st, 16);
        sm[r] += __expf(sinkv - mx[r]);      // sink column in the denominator
        sm[r] = 1.0f / sm[r];
    }

    // ---- write normalized P to LDS (bf16, [query][key]) ----
    unsigned short* Pw = Ps[wave];
    #pragma unroll
    for (int kt = 0; kt < 9; ++kt) {
        #pragma unroll
        for (int r = 0; r < 8; ++r) {
            const int m = r + 8 * half;
            Pw[m * VLD + kt * 16 + lr] = f2bf(sc[kt][r] * sm[r]);
        }
    }
    #pragma unroll
    for (int r = 0; r < 8; ++r) {            // zero pad keys 144..159
        const int m = r + 8 * half;
        Pw[m * VLD + 144 + lr] = 0;
    }
    __syncthreads();

    // ---- PV: out(16x64) = P(16x160) * V(160x64) ----
    v8f out[4];
    #pragma unroll
    for (int t = 0; t < 4; ++t) out[t] = (v8f){0.f,0.f,0.f,0.f,0.f,0.f,0.f,0.f};
    #pragma unroll
    for (int c = 0; c < 5; ++c) {
        Frag16 ap;
        ap.q[0] = *(const uint4*)&Pw[lr * VLD + c * 32 + 8 * half];
        ap.q[1] = *(const uint4*)&Pw[lr * VLD + c * 32 + 16 + 8 * half];
        #pragma unroll
        for (int nt = 0; nt < 4; ++nt) {
            Frag16 bv;  // B elem (k=key, n=hd): fixed hd per lane, keys contiguous
            const int n = nt * 16 + lr;
            bv.q[0] = *(const uint4*)&Vt[n * VLD + c * 32 + 8 * half];
            bv.q[1] = *(const uint4*)&Vt[n * VLD + c * 32 + 16 + 8 * half];
            out[nt] = __builtin_amdgcn_wmma_f32_16x16x32_bf16(
                false, ap.v, false, bv.v, (short)0, out[nt], false, false);
        }
    }

    // ---- store attention output (bf16) ----
    #pragma unroll
    for (int nt = 0; nt < 4; ++nt) {
        #pragma unroll
        for (int r = 0; r < 8; ++r) {
            const int m   = r + 8 * half;
            const int row = qb * 16 + m;
            const int col = h * HD + nt * 16 + lr;
            O[(size_t)row * (NH * HD) + col] = f2bf(out[nt][r]);
        }
    }
}

// ---------------------------------------------------------------------------
// host launcher
// ---------------------------------------------------------------------------
extern "C" void kernel_launch(void* const* d_in, const int* in_sizes, int n_in,
                              void* d_out, int out_size, void* d_ws, size_t ws_size,
                              hipStream_t stream)
{
    (void)in_sizes; (void)n_in; (void)out_size; (void)ws_size;
    const float* x     = (const float*)d_in[0];
    const float* wqkv  = (const float*)d_in[1];
    const float* bqkv  = (const float*)d_in[2];
    const float* wout  = (const float*)d_in[3];
    const float* bout  = (const float*)d_in[4];
    const float* sinks = (const float*)d_in[5];
    const float* cosT  = (const float*)d_in[6];
    const float* sinT  = (const float*)d_in[7];
    // d_in[8] (mask) is reproduced analytically in attn_kernel.
    float* out = (float*)d_out;

    // workspace layout (bytes); total ~75.5 MB
    char* ws = (char*)d_ws;
    unsigned short* x_bf     = (unsigned short*)(ws + 0);          //  8 MB
    unsigned short* wqkv_bt  = (unsigned short*)(ws + 8388608);    // 12 MB [N][K]
    unsigned short* wout_bt  = (unsigned short*)(ws + 20971520);   //  8 MB [N][K]
    float*          qkv_f    = (float*)         (ws + 29360128);   // 24 MB
    unsigned short* q_bf     = (unsigned short*)(ws + 54525952);   //  8 MB
    unsigned short* k_bf     = (unsigned short*)(ws + 62914560);   //  2 MB
    unsigned short* v_bf     = (unsigned short*)(ws + 65011712);   //  2 MB
    unsigned short* attn_bf  = (unsigned short*)(ws + 67108864);   //  8 MB

    const int QKV_DIM = HD * (NH + 2 * KVH);   // 3072

    // 1) fp32 -> bf16 conversions (weights transposed to [N][K])
    cvt_f32_bf16_kernel<<<dim3((S_LEN * HID) / 256), dim3(256), 0, stream>>>(
        x, x_bf, S_LEN * HID);
    cvt_transpose_bf16_kernel<<<dim3(QKV_DIM / 32, HID / 32), dim3(256), 0, stream>>>(
        wqkv, wqkv_bt, HID, QKV_DIM);
    cvt_transpose_bf16_kernel<<<dim3(HID / 32, (NH * HD) / 32), dim3(256), 0, stream>>>(
        wout, wout_bt, NH * HD, HID);

    // 2) QKV GEMM: [2048,2048] x [2048,3072] + bias -> fp32
    gemm_bf16_kernel<<<dim3(QKV_DIM / GBN, S_LEN / GBM), dim3(256), 0, stream>>>(
        x_bf, wqkv_bt, bqkv, qkv_f, S_LEN, QKV_DIM, HID);

    // 3) RoPE + split into q/k/v (bf16); bias already added by GEMM
    rope_split_kernel<<<dim3(S_LEN), dim3(256), 0, stream>>>(
        qkv_f, cosT, sinT, q_bf, k_bf, v_bf);

    // 4) sliding-window GQA attention with sinks (TDM-staged K window)
    attn_kernel<<<dim3(S_LEN / 16, KVH), dim3(128), 0, stream>>>(
        q_bf, k_bf, v_bf, sinks, attn_bf);

    // 5) output GEMM: [2048,2048] x [2048,2048] + bias -> fp32 d_out
    gemm_bf16_kernel<<<dim3(HID / GBN, S_LEN / GBM), dim3(256), 0, stream>>>(
        attn_bf, wout_bt, bout, out, S_LEN, HID, NH * HD);
}
